// AttModel_223338300114
// MI455X (gfx1250) — matrix operand order repaired and checked by reading, not compiled, not run on hardware
//
#include <hip/hip_runtime.h>

// Problem constants
#define BB 32
#define NN 1024
#define DD 128
#define HH 128

typedef __attribute__((ext_vector_type(16))) _Float16 v16h;
typedef __attribute__((ext_vector_type(8)))  _Float16 v8h;
typedef __attribute__((ext_vector_type(2)))  _Float16 v2h;
typedef __attribute__((ext_vector_type(8)))  float    v8f;
typedef int v4i_vs __attribute__((vector_size(4 * sizeof(int))));

static __device__ __forceinline__ v8f wmma_f16(v16h a, v16h b, v8f c) {
  // D = A(16x32 f16) x B(32x16 f16) + C(16x16 f32)
  return __builtin_amdgcn_wmma_f32_16x16x32_f16(false, a, false, b, (short)0, c,
                                                false, false);
}

// --- CDNA5 async global->LDS copy (ASYNCcnt-tracked) -----------------------
// Probe-confirmed signature: (v4i AS1* src, v4i AS3* dst, imm offset, imm cpol)
static __device__ __forceinline__ void async_lds_b128(_Float16* ldst,
                                                      const _Float16* gsrc) {
#if __has_builtin(__builtin_amdgcn_global_load_async_to_lds_b128)
  __builtin_amdgcn_global_load_async_to_lds_b128(
      (__attribute__((address_space(1))) v4i_vs*)(void*)gsrc,
      (__attribute__((address_space(3))) v4i_vs*)(void*)ldst, 0, 0);
#else
  asm volatile("global_load_async_to_lds_b128 %0, %1, off"
               :
               : "v"((__attribute__((address_space(3))) void*)(void*)ldst),
                 "v"(gsrc)
               : "memory");
#endif
}

#if __has_builtin(__builtin_amdgcn_s_wait_asynccnt)
#define WAIT_ASYNC(n) __builtin_amdgcn_s_wait_asynccnt(n)
#else
#define WAIT_ASYNC(n) asm volatile("s_wait_asynccnt %0" ::"i"(n) : "memory")
#endif

// Wave-internal LDS ordering (per-wave scratch: no block barrier needed)
#define WAVE_LDS_FENCE() asm volatile("s_wait_dscnt 0x0" ::: "memory")

// ---------------------------------------------------------------------------
// Kernel 1: fused q/k/v projection. out_w = relu(x @ W_w + b_w), f16.
// x A-fragments loaded once; the three W tiles are staged transposed in LDS
// in turn so each WMMA B-fragment is two contiguous 16B LDS reads per lane.
// ---------------------------------------------------------------------------
__global__ __launch_bounds__(256)
void qkv_proj_kernel(const float* __restrict__ x,
                     const float* __restrict__ Wq, const float* __restrict__ bq,
                     const float* __restrict__ Wk, const float* __restrict__ bk,
                     const float* __restrict__ Wv, const float* __restrict__ bv,
                     _Float16* __restrict__ qo, _Float16* __restrict__ ko,
                     _Float16* __restrict__ vo) {
  __shared__ _Float16 Wt[HH][DD + 8];  // Wt[n][k] = W[k][n], padded rows

  const int tid  = threadIdx.x;
  const int wave = tid >> 5;
  const int lane = tid & 31;
  const int m    = lane & 15;
  const int hi   = lane >> 4;
  const int row0 = (blockIdx.x * 8 + wave) * 16;  // tile row in [B*N]

  // A fragments: x row (f32 -> f16), 4 K-steps of 32. Loaded ONCE for all 3 W.
  // 16-bit A layout: halfs 0..7 = K 8*hi..+7 ; halfs 8..15 = K 16+8*hi..+7
  const float* xrow = x + (size_t)(row0 + m) * DD;
  v16h a[4];
#pragma unroll
  for (int s = 0; s < 4; ++s) {
    const int k0 = 32 * s + 8 * hi;
#pragma unroll
    for (int t = 0; t < 8; ++t) {
      a[s][t]     = (_Float16)xrow[k0 + t];
      a[s][8 + t] = (_Float16)xrow[k0 + 16 + t];
    }
  }

  const float* Ws[3] = {Wq, Wk, Wv};
  const float* bs[3] = {bq, bk, bv};
  _Float16*    os[3] = {qo, ko, vo};

#pragma unroll
  for (int w = 0; w < 3; ++w) {
    if (w) __syncthreads();  // previous round's Wt readers done
    for (int idx = tid; idx < DD * HH; idx += 256) {
      const int kk = idx >> 7;   // DIN (row of W)
      const int nn = idx & 127;  // HID (col of W)
      Wt[nn][kk] = (_Float16)Ws[w][idx];
    }
    __syncthreads();

#pragma unroll
    for (int nt = 0; nt < 8; ++nt) {
      const int h = nt * 16 + m;
      v8f c = {};
#pragma unroll
      for (int s = 0; s < 4; ++s) {
        const _Float16* wr = &Wt[h][32 * s + 8 * hi];
        v16h bf;
#pragma unroll
        for (int t = 0; t < 16; ++t) bf[t] = wr[t < 8 ? t : t + 8];
        c = wmma_f16(a[s], bf, c);
      }
      const float bb = bs[w][h];
#pragma unroll
      for (int r = 0; r < 8; ++r) {
        float vv = c[r] + bb;
        vv = vv > 0.f ? vv : 0.f;
        os[w][(size_t)(row0 + r + 8 * hi) * HH + h] = (_Float16)vv;
      }
    }
  }
}

// ---------------------------------------------------------------------------
// Kernel 2: fused flash attention. Block = 8 waves = 128 query rows of one
// batch. K chunks staged via double-buffered ASYNC global->LDS DMA; V chunks
// software-pipelined through registers and committed transposed to LDS.
// Mask streamed once with NT hints; softmax f32; O accumulated via WMMA.
// ---------------------------------------------------------------------------
__global__ __launch_bounds__(256)
void flash_attn_kernel(const _Float16* __restrict__ q,
                       const _Float16* __restrict__ k,
                       const _Float16* __restrict__ v,
                       const float* __restrict__ mask,
                       float* __restrict__ out) {
  __shared__ _Float16 Kc[2][32][HH + 8];  // K chunk  [buf][j_local][h]
  __shared__ _Float16 Vt[2][HH][40];      // V chunk transposed [buf][h][j_local]
  __shared__ _Float16 Pb[8][16][40];      // per-wave P scratch (D->A relayout)

  const int tid  = threadIdx.x;
  const int wave = tid >> 5;
  const int lane = tid & 31;
  const int m    = lane & 15;
  const int hi   = lane >> 4;
  const int b    = blockIdx.x >> 3;
  const int i0   = (blockIdx.x & 7) * 128 + wave * 16;

  const _Float16* kbase = k + (size_t)b * NN * HH;
  const _Float16* vbase = v + (size_t)b * NN * HH;
  const float*    mbase = mask + (size_t)b * NN * NN;

  // Q A-fragments for this wave's 16 rows (live whole loop)
  const _Float16* qrow = q + ((size_t)b * NN + i0 + m) * HH;
  v16h aq[4];
#pragma unroll
  for (int s = 0; s < 4; ++s) {
    const v8h lo = *(const v8h*)(qrow + 32 * s + 8 * hi);
    const v8h up = *(const v8h*)(qrow + 32 * s + 16 + 8 * hi);
#pragma unroll
    for (int t = 0; t < 8; ++t) { aq[s][t] = lo[t]; aq[s][8 + t] = up[t]; }
  }

  v8f o[8];
#pragma unroll
  for (int nt = 0; nt < 8; ++nt) o[nt] = (v8f){};
  float mrow[8], lrow[8];
#pragma unroll
  for (int r = 0; r < 8; ++r) { mrow[r] = -3.0e38f; lrow[r] = 0.f; }

  // Per-thread staging assignment
  const int jp = (tid >> 4) * 2;      // V: j-pair 0,2,...,30
  const int h0 = (tid & 15) * 8;      // V: 8 h per thread
  const int kjr0 = tid >> 4;          // K xfer 0: row, 16B chunk index
  const int khc0 = tid & 15;
  const int kjr1 = (tid + 256) >> 4;  // K xfer 1
  const int khc1 = tid & 15;

  // ---- Prologue: stage chunk 0 ----
  v8h vreg0 = *(const v8h*)(vbase + (size_t)jp * HH + h0);
  v8h vreg1 = *(const v8h*)(vbase + (size_t)(jp + 1) * HH + h0);
  async_lds_b128(&Kc[0][kjr0][khc0 * 8], kbase + (size_t)kjr0 * HH + khc0 * 8);
  async_lds_b128(&Kc[0][kjr1][khc1 * 8], kbase + (size_t)kjr1 * HH + khc1 * 8);

  for (int c = 0; c < NN / 32; ++c) {
    const int buf = c & 1;
    const int j0  = c * 32;

    __syncthreads();  // barrier X: prior readers of buf (iter c-2) and of
                      // buf^1 (iter c-1) are done; safe to write LDS

    // Commit V chunk c (registers -> LDS, transposed, packed pair stores)
#pragma unroll
    for (int t = 0; t < 8; ++t) {
      v2h p = {vreg0[t], vreg1[t]};
      *(v2h*)&Vt[buf][h0 + t][jp] = p;
    }

    // Prefetch chunk c+1: V into registers, K via async DMA into other buffer
    if (c + 1 < NN / 32) {
      const int jn = j0 + 32;
      vreg0 = *(const v8h*)(vbase + (size_t)(jn + jp) * HH + h0);
      vreg1 = *(const v8h*)(vbase + (size_t)(jn + jp + 1) * HH + h0);
      const int nb = buf ^ 1;
      async_lds_b128(&Kc[nb][kjr0][khc0 * 8],
                     kbase + (size_t)(jn + kjr0) * HH + khc0 * 8);
      async_lds_b128(&Kc[nb][kjr1][khc1 * 8],
                     kbase + (size_t)(jn + kjr1) * HH + khc1 * 8);
      WAIT_ASYNC(2);  // chunk c's 2 async ops retired; c+1's may remain
    } else {
      WAIT_ASYNC(0);
    }
    __syncthreads();  // barrier Y: chunk c fully visible to all waves

    // ---- S = Q * Kc^T : two 16x16 j-tiles, K-dim = H = 128 ----
    v8f st[2];
#pragma unroll
    for (int jt = 0; jt < 2; ++jt) {
      v8f cc = (v8f){};
#pragma unroll
      for (int s = 0; s < 4; ++s) {
        const _Float16* kr = &Kc[buf][jt * 16 + m][32 * s + 8 * hi];
        v16h bf;
#pragma unroll
        for (int t = 0; t < 16; ++t) bf[t] = kr[t < 8 ? t : t + 8];
        cc = wmma_f16(aq[s], bf, cc);
      }
      st[jt] = cc;
    }

    // ---- Mask + online softmax (row r+8*hi lives in one 16-lane group) ----
    float sc[8];
#pragma unroll
    for (int r = 0; r < 8; ++r) {
      const float* mp = mbase + (size_t)(i0 + r + 8 * hi) * NN + j0;
      float rv = -3.0e38f;
#pragma unroll
      for (int jt = 0; jt < 2; ++jt) {
        const float mk = __builtin_nontemporal_load(mp + jt * 16 + m);
        const float lg = st[jt][r] * mk - 9.0e15f * (1.0f - mk);
        st[jt][r] = lg;
        rv = fmaxf(rv, lg);
      }
      rv = fmaxf(rv, __shfl_xor(rv, 1, 32));
      rv = fmaxf(rv, __shfl_xor(rv, 2, 32));
      rv = fmaxf(rv, __shfl_xor(rv, 4, 32));
      rv = fmaxf(rv, __shfl_xor(rv, 8, 32));
      const float mnew = fmaxf(mrow[r], rv);
      sc[r]   = __expf(mrow[r] - mnew);
      mrow[r] = mnew;
      float ps = 0.f;
#pragma unroll
      for (int jt = 0; jt < 2; ++jt) {
        const float p = __expf(st[jt][r] - mnew);
        st[jt][r] = p;
        ps += p;
      }
      ps += __shfl_xor(ps, 1, 32);
      ps += __shfl_xor(ps, 2, 32);
      ps += __shfl_xor(ps, 4, 32);
      ps += __shfl_xor(ps, 8, 32);
      lrow[r] = lrow[r] * sc[r] + ps;
    }

    // Rescale O accumulator
#pragma unroll
    for (int nt = 0; nt < 8; ++nt)
#pragma unroll
      for (int r = 0; r < 8; ++r) o[nt][r] *= sc[r];

    // P (D-layout regs) -> per-wave LDS -> A-layout fragment
#pragma unroll
    for (int jt = 0; jt < 2; ++jt)
#pragma unroll
      for (int r = 0; r < 8; ++r)
        Pb[wave][r + 8 * hi][jt * 16 + m] = (_Float16)st[jt][r];
    WAVE_LDS_FENCE();  // wave-internal only: no block barrier needed

    v16h ap;
    {
      const _Float16* pr = &Pb[wave][m][8 * hi];
#pragma unroll
      for (int t = 0; t < 8; ++t) { ap[t] = pr[t]; ap[8 + t] = pr[16 + t]; }
    }

    // O += P (16x32) x Vchunk (32x128), 8 n-tiles
#pragma unroll
    for (int nt = 0; nt < 8; ++nt) {
      const _Float16* vr = &Vt[buf][nt * 16 + m][8 * hi];
      v16h bf;
#pragma unroll
      for (int t = 0; t < 16; ++t) bf[t] = vr[t < 8 ? t : t + 8];
      o[nt] = wmma_f16(ap, bf, o[nt]);
    }
  }

  // Epilogue: out = O / l
  float* obase = out + ((size_t)b * NN + i0) * HH;
#pragma unroll
  for (int r = 0; r < 8; ++r) {
    const float inv = 1.0f / lrow[r];
#pragma unroll
    for (int nt = 0; nt < 8; ++nt)
      obase[(size_t)(r + 8 * hi) * HH + nt * 16 + m] = o[nt][r] * inv;
  }
}

// ---------------------------------------------------------------------------
extern "C" void kernel_launch(void* const* d_in, const int* in_sizes, int n_in,
                              void* d_out, int out_size, void* d_ws,
                              size_t ws_size, hipStream_t stream) {
  (void)in_sizes; (void)n_in; (void)out_size; (void)ws_size;
  const float* x    = (const float*)d_in[0];
  const float* mask = (const float*)d_in[1];
  const float* Wv   = (const float*)d_in[2];
  const float* bv   = (const float*)d_in[3];
  const float* Wk   = (const float*)d_in[4];
  const float* bk   = (const float*)d_in[5];
  const float* Wq   = (const float*)d_in[6];
  const float* bq   = (const float*)d_in[7];
  float* out = (float*)d_out;

  const size_t qkv_elems = (size_t)BB * NN * HH;  // 4M elems -> 8MB each (f16)
  _Float16* qws = (_Float16*)d_ws;
  _Float16* kws = qws + qkv_elems;
  _Float16* vws = kws + qkv_elems;

  dim3 blk(256);
  dim3 g1((BB * NN) / (16 * 8));  // 2048 tiles / 8 waves per block = 256
  qkv_proj_kernel<<<g1, blk, 0, stream>>>(x, Wq, bq, Wk, bk, Wv, bv,
                                          qws, kws, vws);

  dim3 g2(BB * (NN / 128));  // 256 blocks, 8 waves each
  flash_attn_kernel<<<g2, blk, 0, stream>>>(qws, kws, vws, mask, out);
}
